// EMHSLayer_10462540333069
// MI455X (gfx1250) — matrix operand order repaired and checked by reading, not compiled, not run on hardware
//
#include <hip/hip_runtime.h>

// ---------------------------------------------------------------------------
// Problem constants (from the reference)
// ---------------------------------------------------------------------------
#define NPTS   262144
#define CIN    64
#define FEAT   128
#define OUTC   128
#define LATD   64
#define BSZ    16
#define TOTALV 65536          // BSZ * 16^3
#define EPSBN  1e-5f

// ---------------------------------------------------------------------------
// bf16 / WMMA plumbing (gfx1250, wave32)
// ---------------------------------------------------------------------------
typedef __attribute__((ext_vector_type(16))) __bf16       v16bf;
typedef __attribute__((ext_vector_type(8)))  float        v8f;
typedef __attribute__((ext_vector_type(8)))  unsigned short us8;

union ABfrag {
    v16bf v;
    us8   h[2];
    unsigned short u[16];
};

__device__ __forceinline__ unsigned short f2bf(float f) {
    union { float f; unsigned u; } c; c.f = f;
    unsigned r = c.u + 0x7FFFu + ((c.u >> 16) & 1u);   // round-to-nearest-even
    return (unsigned short)(r >> 16);
}

// ---------------------------------------------------------------------------
// gfx1250 async global->LDS copy (ASYNCcnt-tracked, no VGPR round trip).
// Per-lane: copies 16 bytes from this lane's global address to this lane's
// LDS byte offset. Generic LDS pointers carry the LDS offset in addr[31:0].
// ---------------------------------------------------------------------------
__device__ __forceinline__ void async_to_lds_b128(unsigned lds_off, const void* g) {
    asm volatile("global_load_async_to_lds_b128 %0, %1, off"
                 :: "v"(lds_off), "v"((unsigned long long)(size_t)g)
                 : "memory");
}
__device__ __forceinline__ void wait_asynccnt0() {
    asm volatile("s_wait_asynccnt 0x0" ::: "memory");
}

// ---------------------------------------------------------------------------
// fp32 -> bf16 conversion
// ---------------------------------------------------------------------------
__global__ void k_f32_to_bf16(const float* __restrict__ src,
                              unsigned short* __restrict__ dst, long long n) {
    long long i = (long long)blockIdx.x * blockDim.x + threadIdx.x;
    if (i < n) dst[i] = f2bf(src[i]);
}

// ck[m][c][tap] (fp32)  ->  dst[(m*27+tap)*C + c] (bf16), channel-contiguous for B frags
__global__ void k_convw_to_bf16(const float* __restrict__ ck,
                                unsigned short* __restrict__ dst, int M, int C) {
    long long n = (long long)M * C * 27;
    long long i = (long long)blockIdx.x * blockDim.x + threadIdx.x;
    if (i >= n) return;
    int tap = (int)(i % 27);
    long long t = i / 27;
    int c = (int)(t % C);
    int m = (int)(t / C);
    dst[((long long)m * 27 + tap) * C + c] = f2bf(ck[i]);
}

// ---------------------------------------------------------------------------
// GEMM: Y[N,M] = Xbf[N,K] * Wbf[M,K]^T (+bias) (+=)   -- v_wmma_f32_16x16x32_bf16
// One 16x16 output tile per wave, 8 waves per block.  The block's shared
// X rows (P*16 rows, P = 8/(M/16)) are staged once into LDS via async copy;
// A fragments then come from bank-conflict-free ds_load_b128 (row pad +8).
// ---------------------------------------------------------------------------
__global__ void k_gemm_bf16(const unsigned short* __restrict__ Xbf,
                            const unsigned short* __restrict__ Wbf,
                            const float* __restrict__ bias,
                            float* __restrict__ Y,
                            int Np, int K, int M, int accum) {
    __shared__ unsigned short sX[32 * 136];          // worst case: P=2, K=128
    const int wv     = threadIdx.x >> 5;
    const int lane   = threadIdx.x & 31;
    const int mt     = M >> 4;                       // 4 or 8
    const int P      = 8 / mt;                       // point tiles per block
    const int tid    = blockIdx.x * 8 + wv;
    const int pt     = tid / mt;
    const int ft     = tid - pt * mt;
    const int basept = blockIdx.x * P;
    const int stride = K + 8;                        // pad: conflict-free frags

    // ---- stage P*16 rows of X into LDS (async global->LDS, B128/lane) ----
    {
        const int total = P * 16 * K;
        const unsigned short* gbase = Xbf + ((long long)basept << 4) * K;
        for (int e = threadIdx.x * 8; e < total; e += 256 * 8) {
            int row = e / K, k = e - row * K;
            unsigned lds_off = (unsigned)(size_t)&sX[row * stride + k];
            async_to_lds_b128(lds_off, gbase + (long long)row * K + k);
        }
        wait_asynccnt0();
    }
    __syncthreads();

    const int hi   = lane >> 4;
    const int kbA  = hi << 3;      // lanes 16-31 carry K-subset {8..15, 24..31}
    const int kbB  = hi << 3;      // lanes 16-31 carry K rows {8..15, 24..31}
    const int lp   = wv / mt;      // local point tile inside the block
    const int arow = (lp << 4) + (lane & 15);
    const unsigned short* aBase = &sX[arow * stride];
    const int f    = (ft << 4) + (lane & 15);
    const unsigned short* wrow = Wbf + (long long)f * K;

    v8f acc;
    #pragma unroll
    for (int r = 0; r < 8; ++r) acc[r] = 0.0f;

    for (int k0 = 0; k0 < K; k0 += 32) {
        ABfrag a, b;
        a.h[0] = *(const us8*)(aBase + k0 + kbA);        // ds_load_b128
        a.h[1] = *(const us8*)(aBase + k0 + kbA + 16);
        b.h[0] = *(const us8*)(wrow + k0 + kbB);         // global (L2-resident W)
        b.h[1] = *(const us8*)(wrow + k0 + kbB + 16);
        acc = __builtin_amdgcn_wmma_f32_16x16x32_bf16(
                  false, a.v, false, b.v, (short)0, acc, false, false);
    }

    const float bv = bias ? bias[f] : 0.0f;
    #pragma unroll
    for (int r = 0; r < 8; ++r) {
        int p = (pt << 4) + r + (hi << 3);
        long long o = (long long)p * M + f;
        float v = acc[r] + bv;
        if (accum) Y[o] += v; else Y[o] = v;
    }
}

// ---------------------------------------------------------------------------
// Row softmax over 64 logits, in place; also emits bf16 copy of H.
// ---------------------------------------------------------------------------
__global__ void k_softmax64(float* __restrict__ L,
                            unsigned short* __restrict__ Hbf, int Np) {
    int n = blockIdx.x * blockDim.x + threadIdx.x;
    if (n >= Np) return;
    float* row = L + (long long)n * 64;
    float v[64], mx = -1e30f;
    #pragma unroll
    for (int i = 0; i < 64; ++i) { v[i] = row[i]; mx = fmaxf(mx, v[i]); }
    float s = 0.f;
    #pragma unroll
    for (int i = 0; i < 64; ++i) { v[i] = __expf(v[i] - mx); s += v[i]; }
    float inv = 1.0f / s;
    #pragma unroll
    for (int i = 0; i < 64; ++i) {
        float h = v[i] * inv;
        row[i] = h;
        Hbf[(long long)n * 64 + i] = f2bf(h);
    }
}

// rx[n] = sum_c X[n,c]
__global__ void k_rowsum(const float* __restrict__ X, float* __restrict__ rx,
                         int Np, int C) {
    int n = blockIdx.x * blockDim.x + threadIdx.x;
    if (n >= Np) return;
    const float* row = X + (long long)n * C;
    float s = 0.f;
    for (int c = 0; c < C; ++c) s += row[c];
    rx[n] = s;
}

// s[k] = sum_n H[n,k] * rx[n]
__global__ void k_sreduce(const float* __restrict__ H, const float* __restrict__ rx,
                          float* __restrict__ s, int Np) {
    __shared__ float part[64];
    int t = threadIdx.x;
    if (t < 64) part[t] = 0.f;
    __syncthreads();
    int k = t & 63;
    int g = t >> 6;
    float acc = 0.f;
    for (long long n = (long long)blockIdx.x * 4 + g; n < Np;
         n += (long long)gridDim.x * 4)
        acc += H[n * 64 + k] * rx[n];
    atomicAdd(&part[k], acc);
    __syncthreads();
    if (t < 64) atomicAdd(&s[t], part[t]);
}

// V[m,l] = sum_k attw[m,l,k] * s[k]   ->  bf16 [M,64]
__global__ void k_computeV(const float* __restrict__ attw, const float* __restrict__ s,
                           unsigned short* __restrict__ Vbf, int M) {
    int i = blockIdx.x * blockDim.x + threadIdx.x;
    if (i >= M * 64) return;
    int l = i & 63, m = i >> 6;
    const float* a = attw + ((long long)m * 64 + l) * 64;
    float acc = 0.f;
    for (int k = 0; k < 64; ++k) acc += a[k] * s[k];
    Vbf[(long long)m * 64 + l] = f2bf(acc);
}

// ---------------------------------------------------------------------------
// Scatter-mean voxelization
// ---------------------------------------------------------------------------
__global__ void k_scatter_add(const float* __restrict__ X, const int* __restrict__ cc,
                              float* __restrict__ seg, int Np, int C) {
    long long i = (long long)blockIdx.x * blockDim.x + threadIdx.x;
    long long tot = (long long)Np * C;
    if (i >= tot) return;
    int c = (int)(i % C);
    long long n = i / C;
    atomicAdd(&seg[(long long)cc[n] * C + c], X[i]);
}

__global__ void k_count(const int* __restrict__ cc, float* __restrict__ cnt, int Np) {
    int n = blockIdx.x * blockDim.x + threadIdx.x;
    if (n < Np) atomicAdd(&cnt[cc[n]], 1.0f);
}

// Build bf16 grid, channel-contiguous: gridbf[((b*4096)+v)*C + c2], reproducing
// torch's .view aliasing: flat f = b*C*4096 + c2*4096 + v  <->  f = c'*65536 + u'.
__global__ void k_grid_build(const float* __restrict__ seg, const float* __restrict__ cnt,
                             unsigned short* __restrict__ gridbf, int C) {
    long long o = (long long)blockIdx.x * blockDim.x + threadIdx.x;
    long long tot = (long long)C * TOTALV;
    if (o >= tot) return;
    int c2 = (int)(o % C);
    long long bv = o / C;
    int v = (int)(bv & 4095);
    int b = (int)(bv >> 12);
    long long fidx = ((long long)b * C + c2) * 4096 + v;
    int cp = (int)(fidx >> 16);
    int up = (int)(fidx & 65535);
    float ct = cnt[up];
    float val = seg[(long long)up * C + cp] / fmaxf(ct, 1.0f);
    gridbf[o] = f2bf(val);
}

// ---------------------------------------------------------------------------
// 3x3x3 conv as implicit GEMM over K = 27*C with WMMA (M = 128).
// Block = (b, d, h). The 3*3*18-row halo (shared by all 8 waves) is staged
// once into LDS via async copy (zero rows DS-stored); A fragments then come
// from LDS, B (weights) streams from L2.  convout: (B, M, 4096) + bias.
// ---------------------------------------------------------------------------
__global__ void k_conv3d_wmma(const unsigned short* __restrict__ gridbf,
                              const unsigned short* __restrict__ ckbf,
                              const float* __restrict__ cb,
                              float* __restrict__ convout, int C, int M) {
    __shared__ unsigned short sG[162 * 136];       // worst case C=128 (44 KB)
    const int blk  = blockIdx.x;                   // 0 .. 4095
    const int b    = blk >> 8;
    const int d    = (blk >> 4) & 15;
    const int h    = blk & 15;
    const int ft   = threadIdx.x >> 5;             // 8 waves -> 8 feature tiles
    const int lane = threadIdx.x & 31;
    const int hi   = lane >> 4;
    const int kbA  = hi << 3;
    const int kbB  = hi << 3;
    const int wrow = lane & 15;
    const int f    = (ft << 4) + (lane & 15);
    const int nchunk = C >> 5;
    const int stride = C + 8;

    // ---- stage halo: rows (dz,hz,wz) in [0,3)x[0,3)x[0,18), zero if OOB ----
    {
        const int total = 162 * C;
        for (int e = threadIdx.x * 8; e < total; e += 256 * 8) {
            int ridx = e / C, k = e - ridx * C;
            int dz = ridx / 54, rem = ridx - dz * 54;
            int hz = rem / 18, wz = rem - hz * 18;
            int zd = d + dz - 1, zh = h + hz - 1, zw = wz - 1;
            unsigned short* lp = &sG[ridx * stride + k];
            if ((unsigned)zd < 16u && (unsigned)zh < 16u && (unsigned)zw < 16u) {
                long long arow =
                    ((long long)((b << 12) + (zd << 8) + (zh << 4) + zw)) * C;
                async_to_lds_b128((unsigned)(size_t)lp, gridbf + arow + k);
            } else {
                ((unsigned long long*)lp)[0] = 0ull;   // ds_store zero fill
                ((unsigned long long*)lp)[1] = 0ull;
            }
        }
        wait_asynccnt0();
    }
    __syncthreads();

    v8f acc;
    #pragma unroll
    for (int r = 0; r < 8; ++r) acc[r] = 0.0f;

    for (int tap = 0; tap < 27; ++tap) {
        const int kd = tap / 9, kh = (tap / 3) % 3, kw = tap % 3;
        const unsigned short* aBase =
            &sG[((kd * 3 + kh) * 18 + (wrow + kw)) * stride];
        const unsigned short* bbase = ckbf + ((long long)f * 27 + tap) * C;
        if (tap < 26)
            __builtin_prefetch(ckbf + ((long long)f * 27 + tap + 1) * C, 0, 0);
        for (int ch = 0; ch < nchunk; ++ch) {
            ABfrag a, bm;
            const int c0 = ch << 5;
            a.h[0] = *(const us8*)(aBase + c0 + kbA);     // ds_load_b128
            a.h[1] = *(const us8*)(aBase + c0 + kbA + 16);
            bm.h[0] = *(const us8*)(bbase + c0 + kbB);
            bm.h[1] = *(const us8*)(bbase + c0 + kbB + 16);
            acc = __builtin_amdgcn_wmma_f32_16x16x32_bf16(
                      false, a.v, false, bm.v, (short)0, acc, false, false);
        }
    }

    const float bias = cb[f];
    #pragma unroll
    for (int r = 0; r < 8; ++r) {
        const int wp = r + (hi << 3);
        long long o = (((long long)b * M + f) << 12) + (d << 8) + (h << 4) + wp;
        convout[o] = acc[r] + bias;
    }
}

// outer gather (torch .view faithful): Y[n,m] += convflat[m*65536 + cnc[n]]
__global__ void k_gather_add(const float* __restrict__ convout,
                             const int* __restrict__ cnc,
                             float* __restrict__ Y, int Np, int M) {
    long long i = (long long)blockIdx.x * blockDim.x + threadIdx.x;
    long long tot = (long long)Np * M;
    if (i >= tot) return;
    int m = (int)(i % M);
    long long n = i / M;
    Y[i] += convout[(long long)m * TOTALV + cnc[n]];
}

// ---------------------------------------------------------------------------
// BatchNorm (training mode, biased var) + LeakyReLU(0.2)
// ---------------------------------------------------------------------------
__global__ void k_bn_stats(const float* __restrict__ Y, float* __restrict__ stats,
                           long long Np, int M) {
    __shared__ float ssum[128];
    __shared__ float ssq[128];
    int t = threadIdx.x;
    if (t < M) { ssum[t] = 0.f; ssq[t] = 0.f; }
    __syncthreads();
    long long tot = Np * M;
    for (long long i = (long long)blockIdx.x * blockDim.x + t; i < tot;
         i += (long long)gridDim.x * blockDim.x) {
        float v = Y[i];
        int m = (int)(i % M);
        atomicAdd(&ssum[m], v);
        atomicAdd(&ssq[m], v * v);
    }
    __syncthreads();
    if (t < M) { atomicAdd(&stats[t], ssum[t]); atomicAdd(&stats[M + t], ssq[t]); }
}

__global__ void k_bn_apply_lrelu(const float* __restrict__ Y,
                                 const float* __restrict__ g,
                                 const float* __restrict__ be,
                                 const float* __restrict__ stats,
                                 float* __restrict__ out,
                                 unsigned short* __restrict__ outbf,
                                 long long Np, int M, int addMode) {
    long long i = (long long)blockIdx.x * blockDim.x + threadIdx.x;
    long long tot = Np * M;
    if (i >= tot) return;
    int m = (int)(i % M);
    float mu  = stats[m] / (float)Np;
    float var = stats[M + m] / (float)Np - mu * mu;
    float v = (Y[i] - mu) * rsqrtf(var + EPSBN) * g[m] + be[m];
    v = v > 0.f ? v : 0.2f * v;
    if (addMode) out[i] += v; else out[i] = v;
    if (outbf) outbf[i] = f2bf(v);
}

// ---------------------------------------------------------------------------
// Host orchestration
// ---------------------------------------------------------------------------
struct Ctx {
    float* ybuf;
    float* hlog;
    unsigned short* hbf;
    float* rx;
    float* svec;
    unsigned short* vbf;
    unsigned short* wbf;
    unsigned short* wabf;
    unsigned short* ckbf;
    float* seg;
    float* cnt;
    unsigned short* gridbf;
    float* convf;
    float* bns;
    hipStream_t s;
};

static inline int b1d(long long tot) { return (int)((tot + 255) / 256); }

static void run_elm_layer(const float* Xf, const unsigned short* Xbf,
                          const float* W, const float* bvec,
                          const float* Wa, const float* ba,
                          const float* attw, const float* ckw, const float* cb,
                          const float* g, const float* be,
                          const int* cc, const int* cnc,
                          int C, float* outF, unsigned short* outBF, Ctx& z) {
    const int M = 128;
    // inner = X @ W^T + b
    k_f32_to_bf16<<<b1d((long long)M * C), 256, 0, z.s>>>(W, z.wbf, (long long)M * C);
    k_gemm_bf16<<<(NPTS / 16) * (M / 16) / 8, 256, 0, z.s>>>(Xbf, z.wbf, bvec, z.ybuf,
                                                             NPTS, C, M, 0);
    // H = softmax(X @ Wa^T + ba)
    k_f32_to_bf16<<<b1d((long long)LATD * C), 256, 0, z.s>>>(Wa, z.wabf, (long long)LATD * C);
    k_gemm_bf16<<<(NPTS / 16) * (LATD / 16) / 8, 256, 0, z.s>>>(Xbf, z.wabf, ba, z.hlog,
                                                                NPTS, C, LATD, 0);
    k_softmax64<<<b1d(NPTS), 256, 0, z.s>>>(z.hlog, z.hbf, NPTS);
    // s[k] = sum_n H[n,k]*rowsum(x)[n]  (einsum collapse); V = attw . s
    k_rowsum<<<b1d(NPTS), 256, 0, z.s>>>(Xf, z.rx, NPTS, C);
    hipMemsetAsync(z.svec, 0, 64 * sizeof(float), z.s);
    k_sreduce<<<2048, 256, 0, z.s>>>(z.hlog, z.rx, z.svec, NPTS);
    k_computeV<<<b1d(M * 64), 256, 0, z.s>>>(attw, z.svec, z.vbf, M);
    // inner += H @ V^T
    k_gemm_bf16<<<(NPTS / 16) * (M / 16) / 8, 256, 0, z.s>>>(z.hbf, z.vbf, nullptr,
                                                             z.ybuf, NPTS, LATD, M, 1);
    // scatter-mean -> grid (bf16, channel-contiguous, view-faithful)
    hipMemsetAsync(z.seg, 0, (size_t)TOTALV * C * sizeof(float), z.s);
    hipMemsetAsync(z.cnt, 0, (size_t)TOTALV * sizeof(float), z.s);
    k_scatter_add<<<b1d((long long)NPTS * C), 256, 0, z.s>>>(Xf, cc, z.seg, NPTS, C);
    k_count<<<b1d(NPTS), 256, 0, z.s>>>(cc, z.cnt, NPTS);
    k_grid_build<<<b1d((long long)C * TOTALV), 256, 0, z.s>>>(z.seg, z.cnt, z.gridbf, C);
    // conv3d (implicit GEMM, WMMA, LDS halo)
    k_convw_to_bf16<<<b1d((long long)M * C * 27), 256, 0, z.s>>>(ckw, z.ckbf, M, C);
    k_conv3d_wmma<<<BSZ * 16 * 16, 256, 0, z.s>>>(z.gridbf, z.ckbf, cb, z.convf, C, M);
    // outer gather + add
    k_gather_add<<<b1d((long long)NPTS * M), 256, 0, z.s>>>(z.convf, cnc, z.ybuf, NPTS, M);
    // BN + LeakyReLU
    hipMemsetAsync(z.bns, 0, 2 * M * sizeof(float), z.s);
    k_bn_stats<<<4096, 256, 0, z.s>>>(z.ybuf, z.bns, NPTS, M);
    k_bn_apply_lrelu<<<b1d((long long)NPTS * M), 256, 0, z.s>>>(z.ybuf, g, be, z.bns,
                                                                outF, outBF, NPTS, M, 0);
}

extern "C" void kernel_launch(void* const* d_in, const int* in_sizes, int n_in,
                              void* d_out, int out_size, void* d_ws, size_t ws_size,
                              hipStream_t stream) {
    (void)in_sizes; (void)n_in; (void)out_size; (void)ws_size;
    const float* x     = (const float*)d_in[0];
    const float* W1    = (const float*)d_in[1];
    const float* b1    = (const float*)d_in[2];
    const float* Wa1   = (const float*)d_in[3];
    const float* ba1   = (const float*)d_in[4];
    const float* attw1 = (const float*)d_in[5];
    const float* ck1   = (const float*)d_in[6];
    const float* cb1   = (const float*)d_in[7];
    const float* g1    = (const float*)d_in[8];
    const float* be1   = (const float*)d_in[9];
    const float* W2    = (const float*)d_in[10];
    const float* b2    = (const float*)d_in[11];
    const float* Wa2   = (const float*)d_in[12];
    const float* ba2   = (const float*)d_in[13];
    const float* attw2 = (const float*)d_in[14];
    const float* ck2   = (const float*)d_in[15];
    const float* cb2   = (const float*)d_in[16];
    const float* g2    = (const float*)d_in[17];
    const float* be2   = (const float*)d_in[18];
    const float* Wf    = (const float*)d_in[19];
    const float* bfv   = (const float*)d_in[20];
    const float* gf    = (const float*)d_in[21];
    const float* bef   = (const float*)d_in[22];
    const int*   cc    = (const int*)d_in[23];
    const int*   cnc   = (const int*)d_in[24];
    float* out = (float*)d_out;

    // ---- workspace layout ----
    char* ws = (char*)d_ws;
    size_t off = 0;
    auto take = [&](size_t bytes) -> char* {
        char* p = ws + off;
        off += (bytes + 255) & ~(size_t)255;
        return p;
    };
    unsigned short* XBF    = (unsigned short*)take((size_t)NPTS * CIN * 2);
    Ctx z;
    z.ybuf   = (float*)take((size_t)NPTS * FEAT * 4);
    z.hlog   = (float*)take((size_t)NPTS * LATD * 4);
    z.hbf    = (unsigned short*)take((size_t)NPTS * LATD * 2);
    z.rx     = (float*)take((size_t)NPTS * 4);
    z.svec   = (float*)take(64 * 4);
    z.vbf    = (unsigned short*)take((size_t)FEAT * LATD * 2);
    z.wbf    = (unsigned short*)take((size_t)FEAT * FEAT * 2);
    z.wabf   = (unsigned short*)take((size_t)LATD * FEAT * 2);
    z.ckbf   = (unsigned short*)take((size_t)FEAT * FEAT * 27 * 2);
    z.seg    = (float*)take((size_t)TOTALV * FEAT * 4);
    z.cnt    = (float*)take((size_t)TOTALV * 4);
    z.gridbf = (unsigned short*)take((size_t)FEAT * TOTALV * 2);
    z.convf  = (float*)take((size_t)FEAT * TOTALV * 4);
    float*          H1  = (float*)take((size_t)NPTS * FEAT * 4);
    unsigned short* H1B = (unsigned short*)take((size_t)NPTS * FEAT * 2);
    z.bns    = (float*)take(2 * FEAT * 4);
    z.s = stream;

    // x -> bf16 (used by layer 1 GEMMs and the residual branch)
    k_f32_to_bf16<<<b1d((long long)NPTS * CIN), 256, 0, stream>>>(
        x, XBF, (long long)NPTS * CIN);

    // ---- layer 1: C=64 -> 128 ----
    run_elm_layer(x, XBF, W1, b1, Wa1, ba1, attw1, ck1, cb1, g1, be1,
                  cc, cnc, CIN, H1, H1B, z);

    // ---- layer 2: C=128 -> 128, writes h2 straight into d_out ----
    run_elm_layer(H1, H1B, W2, b2, Wa2, ba2, attw2, ck2, cb2, g2, be2,
                  cc, cnc, FEAT, out, nullptr, z);

    // ---- residual: out += lrelu(BN(x @ Wf^T + bf)) ----
    k_f32_to_bf16<<<b1d((long long)FEAT * CIN), 256, 0, stream>>>(
        Wf, z.wbf, (long long)FEAT * CIN);
    k_gemm_bf16<<<(NPTS / 16) * (FEAT / 16) / 8, 256, 0, stream>>>(
        XBF, z.wbf, bfv, z.ybuf, NPTS, CIN, FEAT, 0);
    hipMemsetAsync(z.bns, 0, 2 * FEAT * sizeof(float), stream);
    k_bn_stats<<<4096, 256, 0, stream>>>(z.ybuf, z.bns, NPTS, FEAT);
    k_bn_apply_lrelu<<<b1d((long long)NPTS * FEAT), 256, 0, stream>>>(
        z.ybuf, gf, bef, z.bns, out, nullptr, NPTS, FEAT, 1);
}
